// CPSFMemcellAutoencoder_51780125720768
// MI455X (gfx1250) — compile-verified, matrix operand order checked by reading, not run on hardware
//
#include <hip/hip_runtime.h>
#include <math.h>

// ---------------------------------------------------------------------------
// CDNA5 (gfx1250) implementation of the CPSF memcell autoencoder.
// All heavy math (convs via implicit im2col, memcell matmuls) goes through a
// single wave32 WMMA GEMM kernel: v_wmma_f32_16x16x32_f16, f32 accumulate,
// fused bias + SiLU/tanh epilogue. Block = 256 threads = 8 waves, tile
// 64Mx128Nx32K; each wave computes a 32x32 slab (2 A-frags x 2 B-frags ->
// 4 WMMA per K-step). Staging is branch-free: clamped unconditional loads +
// value selects, register-batched, with paired-f16 dword stores into LDS.
// ---------------------------------------------------------------------------

typedef __attribute__((ext_vector_type(16))) _Float16 v16h;
typedef __attribute__((ext_vector_type(8)))  float    v8f;

#define TAU_F   5.0f
#define ALPHA_F 0.1f

#define BM 64
#define BN 128
#define BK 32
#define LSTR 36   // LDS row stride in halves (even -> dword-aligned pair ops)

enum { ACT_NONE = 0, ACT_SILU = 1, ACT_TANH = 2 };
enum { GM_ROW = 0, GM_TRN = 1, GM_I2C = 2 };

struct GemmA {
  const float* ptr;
  int lda;                 // modes 0/1
  int Hin, Win, stride, pad, wshift, hshift;   // mode 2 (Hout/Wout are pow2)
};

// Branch-free A-element fetch: the load is unconditional (clamped address),
// out-of-range lowers to a v_cndmask on the value, not an exec branch.
template <int MODE, int CIN>
__device__ __forceinline__ float fetchA(const GemmA& a, int m, int k,
                                        int M, int K) {
  bool ok = (m < M) && (k < K);
  int mc = (m < M) ? m : 0;
  int kc = (k < K) ? k : 0;
  if (MODE == GM_ROW) {
    float v = a.ptr[(size_t)mc * a.lda + kc];
    return ok ? v : 0.0f;
  }
  if (MODE == GM_TRN) {
    float v = a.ptr[(size_t)kc * a.lda + mc];
    return ok ? v : 0.0f;
  }
  // im2col: k = (kh*3+kw)*CIN + ci (ci fastest), m = (b*Hout+oh)*Wout+ow,
  // Hout/Wout powers of two; CIN is a template constant -> magic-multiply.
  int ci = kc % CIN;
  int kk = kc / CIN;
  int kw = kk % 3, kh = kk / 3;
  int ow = mc & ((1 << a.wshift) - 1);
  int t  = mc >> a.wshift;
  int oh = t & ((1 << a.hshift) - 1);
  int b  = t >> a.hshift;
  int ih = oh * a.stride - a.pad + kh;
  int iw = ow * a.stride - a.pad + kw;
  ok = ok && (ih >= 0) && (ih < a.Hin) && (iw >= 0) && (iw < a.Win);
  int ihc = ih < 0 ? 0 : (ih >= a.Hin ? a.Hin - 1 : ih);
  int iwc = iw < 0 ? 0 : (iw >= a.Win ? a.Win - 1 : iw);
  float v = a.ptr[(((size_t)b * a.Hin + ihc) * a.Win + iwc) * CIN + ci];
  return ok ? v : 0.0f;
}

__device__ __forceinline__ float fetchB(const float* __restrict__ Bm, int ldb,
                                        int k, int n, int K, int N) {
  bool ok = (k < K) && (n < N);
  int kc = (k < K) ? k : 0;
  int nc = (n < N) ? n : 0;
  float v = Bm[(size_t)kc * ldb + nc];
  return ok ? v : 0.0f;
}

__device__ __forceinline__ float act_apply(float x, int act) {
  if (act == ACT_SILU) return x / (1.0f + __expf(-x));
  if (act == ACT_TANH) return tanhf(x);
  return x;
}

union Frag { v16h v; unsigned u[8]; };
union HalfPack { _Float16 h[2]; unsigned u; };

#define WMMA_F16(A_, B_, C_)                                                   \
  __builtin_amdgcn_wmma_f32_16x16x32_f16(false, (A_), false, (B_), (short)0,   \
                                         (C_), false, false)

template <int MODE, int CIN>
__global__ __launch_bounds__(256)
void gemm_wmma_f16(GemmA A, const float* __restrict__ Bm, int ldb,
                   float* __restrict__ C, int M, int N, int K, int ldc,
                   const float* __restrict__ bias, float scl, int act)
{
  __shared__ _Float16 sA[BM * LSTR];
  __shared__ _Float16 sB[BN * LSTR];
  const int tid  = threadIdx.x;
  const int wave = tid >> 5;
  const int lane = tid & 31;
  const int wm   = wave >> 2;     // 0..1 -> 32-row slab
  const int wn   = wave & 3;      // 0..3 -> 32-col slab
  const int r16  = lane & 15;
  const int half = lane >> 4;
  const int m0 = blockIdx.y * BM;
  const int n0 = blockIdx.x * BN;

  v8f acc00 = {}, acc01 = {}, acc10 = {}, acc11 = {};

  for (int k0 = 0; k0 < K; k0 += BK) {
    // ---- load phase: all global loads issued branch-free, batched ----
    float va[8];
#pragma unroll
    for (int i = 0; i < 4; ++i) {
      int idx = i * 256 + tid;          // 1024 dword slots (64 x 16)
      int mm  = idx >> 4;               // 0..63
      int kp  = (idx & 15) * 2;         // 0,2,...,30
      va[2 * i + 0] = fetchA<MODE, CIN>(A, m0 + mm, k0 + kp + 0, M, K);
      va[2 * i + 1] = fetchA<MODE, CIN>(A, m0 + mm, k0 + kp + 1, M, K);
    }
    float vb[16];
#pragma unroll
    for (int i = 0; i < 8; ++i) {
      int idx = i * 256 + tid;          // 2048 dword slots (128 x 16)
      int nn  = idx & 127;
      int kp  = (idx >> 7) * 2;
      vb[2 * i + 0] = fetchB(Bm, ldb, k0 + kp + 0, n0 + nn, K, N);
      vb[2 * i + 1] = fetchB(Bm, ldb, k0 + kp + 1, n0 + nn, K, N);
    }
    if (k0 + BK < K)
      __builtin_prefetch(&Bm[(size_t)(k0 + BK) * ldb + n0], 0, 1);

    // ---- store phase: pack two f16 per dword, single b32 DS store ----
#pragma unroll
    for (int i = 0; i < 4; ++i) {
      int idx = i * 256 + tid;
      int mm  = idx >> 4;
      int kp  = (idx & 15) * 2;
      HalfPack p;
      p.h[0] = (_Float16)va[2 * i + 0];
      p.h[1] = (_Float16)va[2 * i + 1];
      *(unsigned*)&sA[mm * LSTR + kp] = p.u;
    }
#pragma unroll
    for (int i = 0; i < 8; ++i) {
      int idx = i * 256 + tid;
      int nn  = idx & 127;
      int kp  = (idx >> 7) * 2;
      HalfPack p;
      p.h[0] = (_Float16)vb[2 * i + 0];
      p.h[1] = (_Float16)vb[2 * i + 1];
      *(unsigned*)&sB[nn * LSTR + kp] = p.u;
    }
    __syncthreads();

    // fragment gather per ISA 16-bit layout:
    //   lanes 0-15:  K pairs {0,1}..{6,7} then {16,17}..{22,23}
    //   lanes 16-31: K pairs {8,9}..{14,15} then {24,25}..{30,31}
    Frag a0, a1, b0, b1;
    const int mA = wm * 32 + r16;
    const int nB = wn * 32 + r16;
#pragma unroll
    for (int i = 0; i < 4; ++i) {
      int kk = half * 8 + 2 * i;
      a0.u[i]     = *(const unsigned*)&sA[mA * LSTR + kk];
      a0.u[4 + i] = *(const unsigned*)&sA[mA * LSTR + kk + 16];
      a1.u[i]     = *(const unsigned*)&sA[(mA + 16) * LSTR + kk];
      a1.u[4 + i] = *(const unsigned*)&sA[(mA + 16) * LSTR + kk + 16];
      b0.u[i]     = *(const unsigned*)&sB[nB * LSTR + kk];
      b0.u[4 + i] = *(const unsigned*)&sB[nB * LSTR + kk + 16];
      b1.u[i]     = *(const unsigned*)&sB[(nB + 16) * LSTR + kk];
      b1.u[4 + i] = *(const unsigned*)&sB[(nB + 16) * LSTR + kk + 16];
    }
    acc00 = WMMA_F16(a0.v, b0.v, acc00);
    acc01 = WMMA_F16(a0.v, b1.v, acc01);
    acc10 = WMMA_F16(a1.v, b0.v, acc10);
    acc11 = WMMA_F16(a1.v, b1.v, acc11);
    __syncthreads();
  }

  // C layout: VGPR r -> M = r + 8*half; N = lane&15
  const int nA = n0 + wn * 32 + r16;
  float bv0 = 0.0f, bv1 = 0.0f;
  if (bias) {
    if (nA < N)      bv0 = bias[nA];
    if (nA + 16 < N) bv1 = bias[nA + 16];
  }
#pragma unroll
  for (int r = 0; r < 8; ++r) {
    int m  = m0 + wm * 32 + half * 8 + r;
    int m2 = m + 16;
    if (nA < N) {
      if (m < M)  C[(size_t)m  * ldc + nA] = act_apply(acc00[r] * scl + bv0, act);
      if (m2 < M) C[(size_t)m2 * ldc + nA] = act_apply(acc10[r] * scl + bv0, act);
    }
    if (nA + 16 < N) {
      if (m < M)  C[(size_t)m  * ldc + nA + 16] = act_apply(acc01[r] * scl + bv1, act);
      if (m2 < M) C[(size_t)m2 * ldc + nA + 16] = act_apply(acc11[r] * scl + bv1, act);
    }
  }
}

// ---------------------------------------------------------------------------
// Support kernels
// ---------------------------------------------------------------------------

__global__ void k_nchw_to_tc(const float* __restrict__ x, float* __restrict__ o,
                             int B, int C, int H, int W) {
  int i = blockIdx.x * blockDim.x + threadIdx.x;
  int n = B * C * H * W;
  if (i >= n) return;
  int w = i % W;
  int h = (i / W) % H;
  int c = (i / (W * H)) % C;
  int b = i / (W * H * C);
  o[(((size_t)b * H + h) * W + w) * C + c] = x[i];
}

// w[Co,Ci,3,3] -> Wm[((kh*3+kw)*Ci+ci) * Co + co]
__global__ void k_pack_conv_w(const float* __restrict__ w, float* __restrict__ o,
                              int Co, int Ci) {
  int i = blockIdx.x * blockDim.x + threadIdx.x;
  int K = Ci * 9;
  if (i >= K * Co) return;
  int co = i % Co;
  int k  = i / Co;
  int ci = k % Ci;
  int kk = k / Ci;
  int kw = kk % 3, kh = kk / 3;
  o[i] = w[(((size_t)co * Ci + ci) * 3 + kh) * 3 + kw];
}

__global__ void k_transpose(const float* __restrict__ in, float* __restrict__ o,
                            int R, int Cc) {
  int i = blockIdx.x * blockDim.x + threadIdx.x;
  if (i >= R * Cc) return;
  int r = i / Cc, c = i % Cc;
  o[(size_t)c * R + r] = in[i];
}

__global__ void k_rowsumsq(const float* __restrict__ z, float* __restrict__ o,
                           int R, int Cc) {
  int r = blockIdx.x * blockDim.x + threadIdx.x;
  if (r >= R) return;
  float s = 0.0f;
  for (int k = 0; k < Cc; ++k) { float v = z[(size_t)r * Cc + k]; s += v * v; }
  o[r] = s;
}

// softmax over M=128 slots of -TAU*(|z|^2 - 2 z.zhat + |zhat|^2); |z|^2 is a
// per-row constant and cancels. One wave32 per row, 8 rows per block.
__global__ __launch_bounds__(256)
void k_softmax_w(const float* __restrict__ S, const float* __restrict__ zh2,
                 float* __restrict__ W, int T) {
  int row  = blockIdx.x * 8 + (threadIdx.x >> 5);
  int lane = threadIdx.x & 31;
  if (row >= T) return;
  const float* s = S + (size_t)row * 128;
  float l[4];
  float mx = -1e30f;
#pragma unroll
  for (int i = 0; i < 4; ++i) {
    int m = lane + 32 * i;
    float v = -TAU_F * (zh2[m] - 2.0f * s[m]);
    l[i] = v;
    mx = fmaxf(mx, v);
  }
#pragma unroll
  for (int off = 16; off > 0; off >>= 1) mx = fmaxf(mx, __shfl_xor(mx, off, 32));
  float sum = 0.0f;
#pragma unroll
  for (int i = 0; i < 4; ++i) { l[i] = __expf(l[i] - mx); sum += l[i]; }
#pragma unroll
  for (int off = 16; off > 0; off >>= 1) sum += __shfl_xor(sum, off, 32);
  float inv = 1.0f / sum;
#pragma unroll
  for (int i = 0; i < 4; ++i)
    W[(size_t)row * 128 + lane + 32 * i] = l[i] * inv;
}

__global__ void k_ew_sub(const float* __restrict__ a, const float* __restrict__ b,
                         float* __restrict__ o, int n) {
  int i = blockIdx.x * blockDim.x + threadIdx.x;
  if (i < n) o[i] = a[i] - b[i];
}

__global__ void k_ew_add(const float* __restrict__ a, const float* __restrict__ b,
                         float* __restrict__ o, int n) {
  int i = blockIdx.x * blockDim.x + threadIdx.x;
  if (i < n) o[i] = a[i] + b[i];
}

// batchnorm stats (biased variance, matching jnp.var ddof=0); one block/channel
__global__ __launch_bounds__(256)
void k_bn_stats(const float* __restrict__ X, int T, int C,
                const float* __restrict__ g, const float* __restrict__ b,
                float* __restrict__ scale, float* __restrict__ shift) {
  __shared__ float s1[256], s2[256];
  int c = blockIdx.x;
  float a0 = 0.0f, a1 = 0.0f;
  for (int t = threadIdx.x; t < T; t += 256) {
    float v = X[(size_t)t * C + c];
    a0 += v; a1 += v * v;
  }
  s1[threadIdx.x] = a0; s2[threadIdx.x] = a1;
  __syncthreads();
  for (int sft = 128; sft > 0; sft >>= 1) {
    if (threadIdx.x < sft) {
      s1[threadIdx.x] += s1[threadIdx.x + sft];
      s2[threadIdx.x] += s2[threadIdx.x + sft];
    }
    __syncthreads();
  }
  if (threadIdx.x == 0) {
    float mean = s1[0] / (float)T;
    float var  = s2[0] / (float)T - mean * mean;
    float inv  = rsqrtf(var + 1e-5f);
    float sc   = g[c] * inv;
    scale[c] = sc;
    shift[c] = b[c] - mean * sc;
  }
}

__global__ void k_bn_apply(float* __restrict__ X, int T, int C,
                           const float* __restrict__ scale,
                           const float* __restrict__ shift) {
  int i = blockIdx.x * blockDim.x + threadIdx.x;
  if (i >= T * C) return;
  int c = i % C;
  X[i] = X[i] * scale[c] + shift[c];
}

// ConvTranspose2d 128->3, k=4, stride=2, pad=1, fused SiLU.
// xin token-major [4*128*128, 128]; wt [128,3,4,4]; y token-major [4*256*256, 3]
__global__ __launch_bounds__(256)
void k_deconv_silu(const float* __restrict__ xin, const float* __restrict__ wt,
                   const float* __restrict__ bt, float* __restrict__ y) {
  __shared__ float sw[128 * 3 * 16];
  for (int i = threadIdx.x; i < 128 * 3 * 16; i += 256) sw[i] = wt[i];
  __syncthreads();
  int idx = blockIdx.x * 256 + threadIdx.x;
  if (idx >= 4 * 256 * 256) return;
  int w = idx % 256;
  int h = (idx / 256) % 256;
  int b = idx / (256 * 256);
  float a0 = bt[0], a1 = bt[1], a2 = bt[2];
#pragma unroll
  for (int kh = 0; kh < 4; ++kh) {
    int hh = h + 1 - kh;
    if (hh < 0 || (hh & 1)) continue;
    int ih = hh >> 1;
    if (ih >= 128) continue;
#pragma unroll
    for (int kw = 0; kw < 4; ++kw) {
      int ww = w + 1 - kw;
      if (ww < 0 || (ww & 1)) continue;
      int iw = ww >> 1;
      if (iw >= 128) continue;
      const float* xp = xin + (((size_t)b * 128 + ih) * 128 + iw) * 128;
      const float* wp = sw + (kh * 4 + kw);
      for (int ci = 0; ci < 128; ++ci) {
        float xv = xp[ci];
        a0 += xv * wp[(ci * 3 + 0) * 16];
        a1 += xv * wp[(ci * 3 + 1) * 16];
        a2 += xv * wp[(ci * 3 + 2) * 16];
      }
    }
  }
  size_t o = (size_t)idx * 3;
  y[o + 0] = a0 / (1.0f + __expf(-a0));
  y[o + 1] = a1 / (1.0f + __expf(-a1));
  y[o + 2] = a2 / (1.0f + __expf(-a2));
}

// final 3x3 conv 3->3 + SiLU, token-major in, NCHW out (matches reference out)
__global__ __launch_bounds__(256)
void k_conv3_silu(const float* __restrict__ y1, const float* __restrict__ wc,
                  const float* __restrict__ bc, float* __restrict__ out) {
  __shared__ float sw[81];
  if (threadIdx.x < 81) sw[threadIdx.x] = wc[threadIdx.x];
  __syncthreads();
  int idx = blockIdx.x * 256 + threadIdx.x;
  if (idx >= 4 * 3 * 256 * 256) return;
  int w  = idx % 256;
  int h  = (idx / 256) % 256;
  int co = (idx / 65536) % 3;
  int b  = idx / (3 * 65536);
  float acc = bc[co];
  for (int kh = 0; kh < 3; ++kh) {
    int ih = h - 1 + kh;
    if (ih < 0 || ih >= 256) continue;
    for (int kw = 0; kw < 3; ++kw) {
      int iw = w - 1 + kw;
      if (iw < 0 || iw >= 256) continue;
      const float* p = y1 + (((size_t)b * 256 + ih) * 256 + iw) * 3;
      for (int ci = 0; ci < 3; ++ci)
        acc += p[ci] * sw[((co * 3 + ci) * 3 + kh) * 3 + kw];
    }
  }
  out[idx] = acc / (1.0f + __expf(-acc));
}

// ---------------------------------------------------------------------------
// Host orchestration
// ---------------------------------------------------------------------------

static GemmA ARow(const float* p, int lda) {
  GemmA a{}; a.ptr = p; a.lda = lda; return a;
}
static GemmA ATrn(const float* p, int lda) {
  GemmA a{}; a.ptr = p; a.lda = lda; return a;
}
static GemmA AI2C(const float* p, int Hin, int Win, int Hout, int Wout, int s) {
  GemmA a{}; a.ptr = p; a.Hin = Hin; a.Win = Win; a.stride = s; a.pad = 1;
  int ws = 0; while ((1 << ws) < Wout) ++ws;
  int hs = 0; while ((1 << hs) < Hout) ++hs;
  a.wshift = ws; a.hshift = hs;
  return a;
}

extern "C" void kernel_launch(void* const* d_in, const int* in_sizes, int n_in,
                              void* d_out, int out_size, void* d_ws, size_t ws_size,
                              hipStream_t stream) {
  (void)in_sizes; (void)n_in; (void)out_size; (void)ws_size;

  const float* x       = (const float*)d_in[0];
  const float* e0n_w1  = (const float*)d_in[1];
  const float* e0n_b1  = (const float*)d_in[2];
  const float* e0n_w2  = (const float*)d_in[3];
  const float* e0n_b2  = (const float*)d_in[4];
  const float* e0s_w1  = (const float*)d_in[5];
  const float* e0s_b1  = (const float*)d_in[6];
  const float* e0s_w2  = (const float*)d_in[7];
  const float* e0s_b2  = (const float*)d_in[8];
  const float* bnn_w1  = (const float*)d_in[9];
  const float* bnn_b1  = (const float*)d_in[10];
  const float* bnn_w2  = (const float*)d_in[11];
  const float* bnn_b2  = (const float*)d_in[12];
  const float* bns_w1  = (const float*)d_in[13];
  const float* bns_b1  = (const float*)d_in[14];
  const float* bns_w2  = (const float*)d_in[15];
  const float* bns_b2  = (const float*)d_in[16];
  const float* d0_wt   = (const float*)d_in[17];
  const float* d0_bt   = (const float*)d_in[18];
  const float* d0_wc   = (const float*)d_in[19];
  const float* d0_bc   = (const float*)d_in[20];
  const float* g0      = (const float*)d_in[21];
  const float* be0     = (const float*)d_in[22];
  const float* gb      = (const float*)d_in[23];
  const float* bb      = (const float*)d_in[24];
  const float* gd      = (const float*)d_in[25];
  const float* bd      = (const float*)d_in[26];
  const float* c0_zhat = (const float*)d_in[27];
  const float* c0_That = (const float*)d_in[28];
  const float* cb_zhat = (const float*)d_in[29];
  const float* cb_That = (const float*)d_in[30];

  const int T0 = 4 * 256 * 256;   // full-res tokens
  const int T1 = 4 * 128 * 128;   // half-res tokens

  // --- workspace bump allocator (fp32) ---
  float* wsf = (float*)d_ws;
  size_t off = 0;
  auto alloc = [&](size_t nf) { float* p = wsf + off; off += nf; return p; };

  float* t_x     = alloc((size_t)T0 * 3);
  float* e0n_t1  = alloc((size_t)T0 * 16);
  float* e0n     = alloc((size_t)T1 * 16);
  float* e0s_t1  = alloc((size_t)T0 * 128);  // reused: bnn_t1 / bns_t1
  float* e0s     = alloc((size_t)T1 * 128);
  float* Sbuf    = alloc((size_t)T1 * 128);
  float* wbuf    = alloc((size_t)T1 * 128);
  float* readb   = alloc((size_t)T1 * 128);
  float* diffb   = alloc((size_t)T1 * 128);
  float* x1      = alloc((size_t)T1 * 128);
  float* d0m     = alloc((size_t)T1 * 128);
  float* bnnb    = alloc((size_t)T1 * 16);
  float* bnsb    = alloc((size_t)T1 * 16);
  float* x2      = alloc((size_t)T1 * 16);
  float* y1      = alloc((size_t)T0 * 3);
  float* delta0  = alloc(128 * 128);
  float* ThatPd0 = alloc(128 * 128);
  float* deltaB  = alloc(128 * 16);
  float* ThatPdB = alloc(128 * 16);
  float* Wz0     = alloc(16 * 128);
  float* Wzb     = alloc(16 * 128);
  float* zh2_0   = alloc(128);
  float* zh2_b   = alloc(128);
  float* scl_c   = alloc(128);
  float* shf_c   = alloc(128);
  float* pw_e0n1 = alloc(27 * 16);
  float* pw_e0n2 = alloc(144 * 16);
  float* pw_e0s1 = alloc(27 * 128);
  float* pw_e0s2 = alloc((size_t)1152 * 128);
  float* pw_bnn1 = alloc((size_t)1152 * 128);
  float* pw_bnn2 = alloc((size_t)1152 * 16);
  float* pw_bns1 = alloc((size_t)1152 * 128);
  float* pw_bns2 = alloc((size_t)1152 * 16);

  auto blocks = [](int n) { return (unsigned)((n + 255) / 256); };

  auto gemm = [&](int mode, int cin, GemmA A, const float* Bm, int ldb, float* C,
                  int M, int N, int K, const float* bias, float scl, int act) {
    dim3 g((unsigned)((N + BN - 1) / BN), (unsigned)((M + BM - 1) / BM));
    if (mode == GM_ROW)
      gemm_wmma_f16<GM_ROW, 1><<<g, 256, 0, stream>>>(A, Bm, ldb, C, M, N, K, N,
                                                      bias, scl, act);
    else if (mode == GM_TRN)
      gemm_wmma_f16<GM_TRN, 1><<<g, 256, 0, stream>>>(A, Bm, ldb, C, M, N, K, N,
                                                      bias, scl, act);
    else if (cin == 3)
      gemm_wmma_f16<GM_I2C, 3><<<g, 256, 0, stream>>>(A, Bm, ldb, C, M, N, K, N,
                                                      bias, scl, act);
    else if (cin == 16)
      gemm_wmma_f16<GM_I2C, 16><<<g, 256, 0, stream>>>(A, Bm, ldb, C, M, N, K, N,
                                                       bias, scl, act);
    else
      gemm_wmma_f16<GM_I2C, 128><<<g, 256, 0, stream>>>(A, Bm, ldb, C, M, N, K, N,
                                                        bias, scl, act);
  };

  // ---- layout + weight prep ----
  k_nchw_to_tc<<<blocks(T0 * 3), 256, 0, stream>>>(x, t_x, 4, 3, 256, 256);
  k_pack_conv_w<<<blocks(27 * 16), 256, 0, stream>>>(e0n_w1, pw_e0n1, 16, 3);
  k_pack_conv_w<<<blocks(144 * 16), 256, 0, stream>>>(e0n_w2, pw_e0n2, 16, 16);
  k_pack_conv_w<<<blocks(27 * 128), 256, 0, stream>>>(e0s_w1, pw_e0s1, 128, 3);
  k_pack_conv_w<<<blocks(1152 * 128), 256, 0, stream>>>(e0s_w2, pw_e0s2, 128, 128);
  k_pack_conv_w<<<blocks(1152 * 128), 256, 0, stream>>>(bnn_w1, pw_bnn1, 128, 128);
  k_pack_conv_w<<<blocks(1152 * 16), 256, 0, stream>>>(bnn_w2, pw_bnn2, 16, 128);
  k_pack_conv_w<<<blocks(1152 * 128), 256, 0, stream>>>(bns_w1, pw_bns1, 128, 128);
  k_pack_conv_w<<<blocks(1152 * 16), 256, 0, stream>>>(bns_w2, pw_bns2, 16, 128);
  k_transpose<<<blocks(128 * 16), 256, 0, stream>>>(c0_zhat, Wz0, 128, 16);
  k_transpose<<<blocks(128 * 16), 256, 0, stream>>>(cb_zhat, Wzb, 128, 16);
  k_rowsumsq<<<1, 128, 0, stream>>>(c0_zhat, zh2_0, 128, 16);
  k_rowsumsq<<<1, 128, 0, stream>>>(cb_zhat, zh2_b, 128, 16);

  // ---- encoder scale 0 ----
  gemm(GM_I2C, 3, AI2C(t_x, 256, 256, 256, 256, 1), pw_e0n1, 16, e0n_t1,
       T0, 16, 27, e0n_b1, 1.0f, ACT_SILU);
  gemm(GM_I2C, 16, AI2C(e0n_t1, 256, 256, 128, 128, 2), pw_e0n2, 16, e0n,
       T1, 16, 144, e0n_b2, 1.0f, ACT_SILU);
  gemm(GM_I2C, 3, AI2C(t_x, 256, 256, 256, 256, 1), pw_e0s1, 128, e0s_t1,
       T0, 128, 27, e0s_b1, 1.0f, ACT_SILU);
  gemm(GM_I2C, 128, AI2C(e0s_t1, 256, 256, 128, 128, 2), pw_e0s2, 128, e0s,
       T1, 128, 1152, e0s_b2, 1.0f, ACT_SILU);

  // ---- memcell 0 write: out0 = w @ (That + alpha * w^T (T* - w@That)) ----
  gemm(GM_ROW, 0, ARow(e0n, 16), Wz0, 128, Sbuf, T1, 128, 16,
       nullptr, 1.0f, ACT_NONE);
  k_softmax_w<<<blocks(T1 * 32), 256, 0, stream>>>(Sbuf, zh2_0, wbuf, T1);
  gemm(GM_ROW, 0, ARow(wbuf, 128), c0_That, 128, readb, T1, 128, 128,
       nullptr, 1.0f, ACT_NONE);
  k_ew_sub<<<blocks(T1 * 128), 256, 0, stream>>>(e0s, readb, diffb, T1 * 128);
  gemm(GM_TRN, 0, ATrn(wbuf, 128), diffb, 128, delta0, 128, 128, T1,
       nullptr, ALPHA_F, ACT_NONE);
  k_ew_add<<<blocks(128 * 128), 256, 0, stream>>>(c0_That, delta0, ThatPd0, 128 * 128);
  gemm(GM_ROW, 0, ARow(wbuf, 128), ThatPd0, 128, x1, T1, 128, 128,
       nullptr, 1.0f, ACT_TANH);
  k_bn_stats<<<128, 256, 0, stream>>>(x1, T1, 128, g0, be0, scl_c, shf_c);
  k_bn_apply<<<blocks(T1 * 128), 256, 0, stream>>>(x1, T1, 128, scl_c, shf_c);

  // ---- bottleneck (reuse big e0s_t1 region) ----
  float* bnn_t1 = e0s_t1;
  float* bns_t1 = e0s_t1 + (size_t)T1 * 128;
  gemm(GM_I2C, 128, AI2C(x1, 128, 128, 128, 128, 1), pw_bnn1, 128, bnn_t1,
       T1, 128, 1152, bnn_b1, 1.0f, ACT_SILU);
  gemm(GM_I2C, 128, AI2C(bnn_t1, 128, 128, 128, 128, 1), pw_bnn2, 16, bnnb,
       T1, 16, 1152, bnn_b2, 1.0f, ACT_SILU);
  gemm(GM_I2C, 128, AI2C(x1, 128, 128, 128, 128, 1), pw_bns1, 128, bns_t1,
       T1, 128, 1152, bns_b1, 1.0f, ACT_SILU);
  gemm(GM_I2C, 128, AI2C(bns_t1, 128, 128, 128, 128, 1), pw_bns2, 16, bnsb,
       T1, 16, 1152, bns_b2, 1.0f, ACT_SILU);

  // ---- memcell b write ----
  gemm(GM_ROW, 0, ARow(bnnb, 16), Wzb, 128, Sbuf, T1, 128, 16,
       nullptr, 1.0f, ACT_NONE);
  k_softmax_w<<<blocks(T1 * 32), 256, 0, stream>>>(Sbuf, zh2_b, wbuf, T1);
  gemm(GM_ROW, 0, ARow(wbuf, 128), cb_That, 16, readb, T1, 16, 128,
       nullptr, 1.0f, ACT_NONE);
  k_ew_sub<<<blocks(T1 * 16), 256, 0, stream>>>(bnsb, readb, diffb, T1 * 16);
  gemm(GM_TRN, 0, ATrn(wbuf, 128), diffb, 16, deltaB, 128, 16, T1,
       nullptr, ALPHA_F, ACT_NONE);
  k_ew_add<<<blocks(128 * 16), 256, 0, stream>>>(cb_That, deltaB, ThatPdB, 128 * 16);
  gemm(GM_ROW, 0, ARow(wbuf, 128), ThatPdB, 16, x2, T1, 16, 128,
       nullptr, 1.0f, ACT_TANH);
  k_bn_stats<<<16, 256, 0, stream>>>(x2, T1, 16, gb, bb, scl_c, shf_c);
  k_bn_apply<<<blocks(T1 * 16), 256, 0, stream>>>(x2, T1, 16, scl_c, shf_c);

  // ---- decoder: read-only recall against cell 0 (uses delta0) ----
  gemm(GM_ROW, 0, ARow(x2, 16), Wz0, 128, Sbuf, T1, 128, 16,
       nullptr, 1.0f, ACT_NONE);
  k_softmax_w<<<blocks(T1 * 32), 256, 0, stream>>>(Sbuf, zh2_0, wbuf, T1);
  gemm(GM_ROW, 0, ARow(wbuf, 128), ThatPd0, 128, d0m, T1, 128, 128,
       nullptr, 1.0f, ACT_TANH);
  k_bn_stats<<<128, 256, 0, stream>>>(d0m, T1, 128, gd, bd, scl_c, shf_c);
  k_bn_apply<<<blocks(T1 * 128), 256, 0, stream>>>(d0m, T1, 128, scl_c, shf_c);

  // ---- deconv + final conv ----
  k_deconv_silu<<<blocks(T0), 256, 0, stream>>>(d0m, d0_wt, d0_bt, y1);
  k_conv3_silu<<<blocks(T0 * 3), 256, 0, stream>>>(y1, d0_wc, d0_bc, (float*)d_out);
}
